// IterativeLSTMClassifier_72086731096831
// MI455X (gfx1250) — matrix operand order repaired and checked by reading, not compiled, not run on hardware
//
#include <hip/hip_runtime.h>
#include <hip/hip_bf16.h>
#include <math.h>

typedef __bf16 bf16;
typedef __attribute__((ext_vector_type(16))) __bf16 v16bf;
typedef __attribute__((ext_vector_type(8)))  float  v8f;
typedef __attribute__((ext_vector_type(4)))  int    v4i;

#define B_   64
#define T_   256
#define E_   300
#define EP   320      // E padded to K multiple of 64
#define H_   512
#define G4   2048     // 4*H
#define BT   16384    // B*T
#define AK   812      // E+H attention K
#define AKP  832      // padded to multiple of 64
#define AN   300      // attention hidden
#define ANP  384      // padded to 3*128
#define PADTOK 1
#define STEP_KC 256   // LDS K-chunk for hx staging (64*256*2 = 32KB)

#if defined(__has_builtin)
#  if __has_builtin(__builtin_amdgcn_global_load_async_to_lds_b128)
#    define USE_ASYNC_LDS 1
#  endif
#endif
#ifndef USE_ASYNC_LDS
#  define USE_ASYNC_LDS 0
#endif

static __device__ __forceinline__ float sigf(float x){ return 1.f/(1.f+__expf(-x)); }

// ---------- WMMA fragment helpers (CDNA5 ISA 7.12.2 layouts, wave32) ----------
// A: 16x32 bf16, lanes 0-15 = rows M holding K{0..7,16..23}, lanes 16-31 K{8..15,24..31}
static __device__ __forceinline__ v16bf loadA(const bf16* A, int lda, int m0, int kk, int lane){
  int half = lane >> 4, mr = lane & 15;
  const bf16* p = A + (size_t)(m0 + mr) * lda + kk + 8 * half;
  v16bf a;
#pragma unroll
  for (int v = 0; v < 8; ++v){
    int kb = (v < 4) ? (2*v) : (16 + 2*(v-4));
    a[2*v]   = p[kb];
    a[2*v+1] = p[kb+1];
  }
  return a;
}
// B: 32x16 bf16 (KxN). Bt stored [N][Kpad] row-major (== torch W[out,in] natural layout),
// lanes 0-15 hold K{0..7,16..23} (N=lane), lanes 16-31 hold K{8..15,24..31} (matches loadA).
static __device__ __forceinline__ v16bf loadBt(const bf16* Bt, int ldk, int n0, int kk, int lane){
  int half = lane >> 4, nc = lane & 15;
  const bf16* p = Bt + (size_t)(n0 + nc) * ldk + kk + 8 * half;
  v16bf b;
#pragma unroll
  for (int e = 0; e < 16; ++e) b[e] = p[(e < 8) ? e : (e + 8)];
  return b;
}
// C/D 16x16 f32: VGPR r -> row m0 + r + 8*half, col n0 + (lane&15)

// ---------- small prep kernels ----------
__global__ void k_convert_pad(const float* src, bf16* dst, int srows, int scols,
                              int drows, int dcols){
  size_t n = (size_t)drows * dcols;
  for (size_t i = (size_t)blockIdx.x*blockDim.x + threadIdx.x; i < n; i += (size_t)gridDim.x*blockDim.x){
    int r = (int)(i / dcols), c = (int)(i % dcols);
    float v = (r < srows && c < scols) ? src[(size_t)r*scols + c] : 0.f;
    dst[i] = (bf16)v;
  }
}
__global__ void k_bsum(const float* a, const float* b, float* o, int n){
  int i = blockIdx.x*blockDim.x + threadIdx.x;
  if (i < n) o[i] = a[i] + b[i];
}
__global__ void k_fpad(const float* a, float* o, int n, int np){
  int i = blockIdx.x*blockDim.x + threadIdx.x;
  if (i < np) o[i] = (i < n) ? a[i] : 0.f;
}
__global__ void k_zero_f(float* p, int n){
  int i = blockIdx.x*blockDim.x + threadIdx.x;
  if (i < n) p[i] = 0.f;
}
__global__ void k_zero_bf(bf16* p, int n){
  int i = blockIdx.x*blockDim.x + threadIdx.x;
  if (i < n) p[i] = (bf16)0.f;
}
__global__ void k_embed(const float* emb, const int* x, bf16* inpb){
  size_t n = (size_t)BT * EP;
  for (size_t i = (size_t)blockIdx.x*blockDim.x + threadIdx.x; i < n; i += (size_t)gridDim.x*blockDim.x){
    int r = (int)(i / EP), k = (int)(i % EP);
    int tok = x[r];
    float v = (k < E_ && tok != PADTOK) ? emb[(size_t)tok*E_ + k] : 0.f;
    inpb[i] = (bf16)v;
  }
}
__global__ void k_lengths(const int* x, int* len){
  int b = blockIdx.x*blockDim.x + threadIdx.x;
  if (b < B_){
    int c = 0;
    for (int t = 0; t < T_; ++t) c += (x[b*T_ + t] != PADTOK) ? 1 : 0;
    len[b] = c;
  }
}
// hx <- last, cx <- 0, hbf <- bf16(last) between iterations
__global__ void k_prep2(const float* lastf, float* hf, float* cf, bf16* hb0){
  int i = blockIdx.x*blockDim.x + threadIdx.x;
  if (i < B_*H_){ float v = lastf[i]; hf[i] = v; cf[i] = 0.f; hb0[i] = (bf16)v; }
}
// att_in = concat(inp[:,:,0:300], last[b]) padded to 832, bf16
__global__ void k_attin(const bf16* inpb, const bf16* lastb, bf16* attin){
  size_t n = (size_t)BT * AKP;
  for (size_t i = (size_t)blockIdx.x*blockDim.x + threadIdx.x; i < n; i += (size_t)gridDim.x*blockDim.x){
    int r = (int)(i / AKP), k = (int)(i % AKP);
    int b = r >> 8;  // r / T_
    bf16 v;
    if (k < E_)       v = inpb[(size_t)r*EP + k];
    else if (k < AK)  v = lastb[(size_t)b*H_ + (k - E_)];
    else              v = (bf16)0.f;
    attin[i] = v;
  }
}
__global__ void k_attreduce(const float* h1, const float* aW2, const float* ab2, float* att){
  int row = blockIdx.x;        // BT rows
  int lane = threadIdx.x;      // 32 threads
  float s = 0.f;
  for (int j = lane; j < AN; j += 32) s += h1[(size_t)row*ANP + j] * aW2[j];
#pragma unroll
  for (int m = 16; m; m >>= 1) s += __shfl_xor(s, m, 32);
  if (lane == 0) att[row] = sigf(s + ab2[0]);
}
__global__ void k_out(const float* lastf, const float* Wout, const float* bout, float* out){
  int tid = threadIdx.x;
  if (tid >= B_*5) return;
  int b = tid / 5, o = tid % 5;
  float s = bout[o];
  for (int k = 0; k < H_; ++k) s += lastf[b*H_ + k] * Wout[o*H_ + k];
  out[tid] = s;
}

// ---------- generic bf16 GEMM: C[M,N] = act(A[M,lda] x Bt[N,ldk]^T + bias[N]) ----------
// block = 256 threads (8 waves), tile 64(M) x 128(N); grid = (M/64, N/128).
// 2x-unrolled ping-pong pipeline: buffer p is refilled while buffer 1-p feeds WMMAs,
// no register-rotation copies. Requires K % 64 == 0 (EP=320, AKP=832 both qualify).
__global__ void __launch_bounds__(256) k_gemm(const bf16* __restrict__ A, int lda,
                                              const bf16* __restrict__ Bt, int ldk,
                                              const float* __restrict__ bias,
                                              float* __restrict__ C, int ldc,
                                              int K, int act){
  int tid = threadIdx.x, lane = tid & 31, w = tid >> 5;
  int m0    = blockIdx.x * 64  + (w & 3) * 16;
  int nbase = blockIdx.y * 128 + (w >> 2) * 64;
  v8f acc[4] = {};
  v16bf aF[2], bF[2][4];
  aF[0] = loadA(A, lda, m0, 0, lane);
#pragma unroll
  for (int j = 0; j < 4; ++j) bF[0][j] = loadBt(Bt, ldk, nbase + j*16, 0, lane);

  for (int kk = 0; kk < K; kk += 64){
    int k1 = kk + 32;                       // always < K (K % 64 == 0)
    aF[1] = loadA(A, lda, m0, k1, lane);
#pragma unroll
    for (int j = 0; j < 4; ++j) bF[1][j] = loadBt(Bt, ldk, nbase + j*16, k1, lane);
#pragma unroll
    for (int j = 0; j < 4; ++j)
      acc[j] = __builtin_amdgcn_wmma_f32_16x16x32_bf16(false, aF[0], false, bF[0][j],
                                                       (short)0, acc[j], false, false);
    int k2 = kk + 64; if (k2 >= K) k2 = 0;  // dummy wrap on final iteration
    aF[0] = loadA(A, lda, m0, k2, lane);
#pragma unroll
    for (int j = 0; j < 4; ++j) bF[0][j] = loadBt(Bt, ldk, nbase + j*16, k2, lane);
#pragma unroll
    for (int j = 0; j < 4; ++j)
      acc[j] = __builtin_amdgcn_wmma_f32_16x16x32_bf16(false, aF[1], false, bF[1][j],
                                                       (short)0, acc[j], false, false);
  }

  int nc = lane & 15, half = lane >> 4;
#pragma unroll
  for (int j = 0; j < 4; ++j){
    int n0 = nbase + j*16;
    float bv = bias ? bias[n0 + nc] : 0.f;
#pragma unroll
    for (int r = 0; r < 8; ++r){
      float v = acc[j][r] + bv;
      if (act == 1) v = tanhf(v);
      C[(size_t)(m0 + r + 8*half)*ldc + n0 + nc] = v;
    }
  }
}

// ---------- one LSTM timestep: gates = xg[:,t,:] + hx @ Whh^T, then cell update ----------
// grid = 32 blocks (16 h-cols each, plus matching i/f/g/o gate chunks), block = 128 thr (4 waves)
__global__ void __launch_bounds__(128) k_step(const bf16* __restrict__ h_in_bf,  // [64][512]
                                              bf16* __restrict__ h_out_bf,       // [64][512]
                                              float* __restrict__ h_f,           // [64][512]
                                              float* __restrict__ c_f,           // [64][512]
                                              const float* __restrict__ xg,      // [BT][2048]
                                              const bf16* __restrict__ Whh,      // [2048][512] bf16
                                              float* __restrict__ lastf,         // [64][512]
                                              const float* __restrict__ att,     // [BT] or null
                                              const int* __restrict__ len,
                                              int t){
  __shared__ bf16 sh[B_ * STEP_KC];
  int tid = threadIdx.x, lane = tid & 31, w = tid >> 5;
  int m0 = w * 16;                 // 4 waves cover M=64
  int hcol0 = blockIdx.x * 16;     // this block's 16 h-columns
  v8f acc[4] = {};                 // i, f, g, o
  v16bf bF[2][4];

  // prime ping buffer with K=0 fragments (overlaps with first LDS stage below)
#pragma unroll
  for (int g = 0; g < 4; ++g) bF[0][g] = loadBt(Whh, H_, g*H_ + hcol0, 0, lane);

  for (int kc = 0; kc < H_; kc += STEP_KC){
    // stage hx[:, kc:kc+KC] into LDS (16B per transfer, async if available)
    for (int i = tid; i < (B_*STEP_KC)/8; i += 128){
      int r = i >> 5, c8 = i & 31;               // STEP_KC/8 == 32
      const bf16* gsrc = h_in_bf + (size_t)r*H_ + kc + c8*8;
      bf16* ldst = sh + r*STEP_KC + c8*8;
#if USE_ASYNC_LDS
      __builtin_amdgcn_global_load_async_to_lds_b128(
          (__attribute__((address_space(1))) v4i*)
              (__attribute__((address_space(1))) void*)(void*)(const void*)gsrc,
          (__attribute__((address_space(3))) v4i*)
              (__attribute__((address_space(3))) void*)(void*)ldst,
          0, 0);
#else
      *(uint4*)ldst = *(const uint4*)gsrc;
#endif
    }
#if USE_ASYNC_LDS
    asm volatile("s_wait_asynccnt 0" ::: "memory");
#endif
    __syncthreads();

    // 2x-unrolled ping-pong K loop; last sub-iteration prefetches next chunk's k=kc+KC
#pragma unroll
    for (int kk = 0; kk < STEP_KC; kk += 64){
      int kg = kc + kk;
#pragma unroll
      for (int g = 0; g < 4; ++g) bF[1][g] = loadBt(Whh, H_, g*H_ + hcol0, kg + 32, lane);
      v16bf a0 = loadA(sh, STEP_KC, m0, kk, lane);
#pragma unroll
      for (int g = 0; g < 4; ++g)
        acc[g] = __builtin_amdgcn_wmma_f32_16x16x32_bf16(false, a0, false, bF[0][g],
                                                         (short)0, acc[g], false, false);
      int kn = kg + 64; if (kn >= H_) kn = 0;   // cross-chunk prefetch / dummy wrap
#pragma unroll
      for (int g = 0; g < 4; ++g) bF[0][g] = loadBt(Whh, H_, g*H_ + hcol0, kn, lane);
      v16bf a1 = loadA(sh, STEP_KC, m0, kk + 32, lane);
#pragma unroll
      for (int g = 0; g < 4; ++g)
        acc[g] = __builtin_amdgcn_wmma_f32_16x16x32_bf16(false, a1, false, bF[1][g],
                                                         (short)0, acc[g], false, false);
    }
    __syncthreads();
  }

  int nc = lane & 15, half = lane >> 4;
  int col = hcol0 + nc;
#pragma unroll
  for (int r = 0; r < 8; ++r){
    int b = m0 + r + 8*half;                     // batch row 0..63
    size_t row = (size_t)b * T_ + t;
    const float* xr = xg + row * G4;
    float gi = acc[0][r] + xr[0*H_ + col];
    float gf = acc[1][r] + xr[1*H_ + col];
    float gg = acc[2][r] + xr[2*H_ + col];
    float go = acc[3][r] + xr[3*H_ + col];
    float i_ = sigf(gi), f_ = sigf(gf), g_ = tanhf(gg), o_ = sigf(go);
    size_t hi = (size_t)b*H_ + col;
    float cprev = c_f[hi];
    float cn = f_*cprev + i_*g_;
    float hn = o_*tanhf(cn);
    if (att){
      float a_ = att[row];
      float hp = h_f[hi];
      hn = a_*hn + (1.f - a_)*hp;
      cn = a_*cn + (1.f - a_)*cprev;
    }
    c_f[hi] = cn;
    h_f[hi] = hn;
    h_out_bf[hi] = (bf16)hn;
    if (t == len[b] - 1) lastf[hi] = hn;
  }
  // keep this block's W_hh slice warm for the next timestep's launch
  __builtin_prefetch(Whh + (size_t)(hcol0 + (tid & 15)) * H_, 0, 3);
}

extern "C" void kernel_launch(void* const* d_in, const int* in_sizes, int n_in,
                              void* d_out, int out_size, void* d_ws, size_t ws_size,
                              hipStream_t stream){
  const float* emb  = (const float*)d_in[0];
  const float* W_ih = (const float*)d_in[1];
  const float* b_ih = (const float*)d_in[2];
  const float* W_hh = (const float*)d_in[3];
  const float* b_hh = (const float*)d_in[4];
  const float* aW1  = (const float*)d_in[5];
  const float* ab1  = (const float*)d_in[6];
  const float* aW2  = (const float*)d_in[7];
  const float* ab2  = (const float*)d_in[8];
  const float* Wout = (const float*)d_in[9];
  const float* bout = (const float*)d_in[10];
  const int*   x    = (const int*)d_in[11];

  size_t off = 0;
  auto carve = [&](size_t bytes)->char*{
    off = (off + 255) & ~(size_t)255;
    char* r = (char*)d_ws + off; off += bytes; return r;
  };
  bf16*  inpb  = (bf16*) carve((size_t)BT*EP*2);
  bf16*  Wihb  = (bf16*) carve((size_t)G4*EP*2);
  bf16*  Whhb  = (bf16*) carve((size_t)G4*H_*2);
  bf16*  aW1b  = (bf16*) carve((size_t)ANP*AKP*2);
  float* ab1p  = (float*)carve((size_t)ANP*4);
  float* bsum  = (float*)carve((size_t)G4*4);
  float* xg    = (float*)carve((size_t)BT*G4*4);
  float* hf    = (float*)carve((size_t)B_*H_*4);
  float* cf    = (float*)carve((size_t)B_*H_*4);
  float* lastf = (float*)carve((size_t)B_*H_*4);
  bf16*  hb0   = (bf16*) carve((size_t)B_*H_*2);
  bf16*  hb1   = (bf16*) carve((size_t)B_*H_*2);
  bf16*  attin = (bf16*) carve((size_t)BT*AKP*2);
  float* h1    = (float*)carve((size_t)BT*ANP*4);
  float* attv  = (float*)carve((size_t)BT*4);
  int*   len   = (int*)  carve((size_t)B_*4);
  bf16*  hb[2] = { hb0, hb1 };
  (void)ws_size; (void)n_in; (void)in_sizes; (void)out_size;

  // weight conversion / padding (bf16, natural [N][K] layouts feed B-fragments directly)
  k_convert_pad<<<512, 256, 0, stream>>>(W_ih, Wihb, G4, E_, G4, EP);
  k_convert_pad<<<512, 256, 0, stream>>>(W_hh, Whhb, G4, H_, G4, H_);
  k_convert_pad<<<512, 256, 0, stream>>>(aW1,  aW1b, AN, AK, ANP, AKP);
  k_bsum<<<(G4+255)/256, 256, 0, stream>>>(b_ih, b_hh, bsum, G4);
  k_fpad<<<(ANP+255)/256, 256, 0, stream>>>(ab1, ab1p, AN, ANP);
  k_embed<<<2048, 256, 0, stream>>>(emb, x, inpb);
  k_lengths<<<1, 64, 0, stream>>>(x, len);
  k_zero_f<<<(B_*H_+255)/256, 256, 0, stream>>>(cf, B_*H_);
  k_zero_f<<<(B_*H_+255)/256, 256, 0, stream>>>(hf, B_*H_);
  k_zero_bf<<<(B_*H_+255)/256, 256, 0, stream>>>(hb0, B_*H_);

  // hoisted input projection: xg = inp @ W_ih^T + (b_ih + b_hh)   [BT x 2048]
  k_gemm<<<dim3(BT/64, G4/128), 256, 0, stream>>>(inpb, EP, Wihb, EP, bsum, xg, G4, EP, 0);

  // iteration 1: plain LSTM scan
  int cur = 0;
  for (int t = 0; t < T_; ++t){
    k_step<<<32, 128, 0, stream>>>(hb[cur], hb[cur^1], hf, cf, xg, Whhb,
                                   lastf, nullptr, len, t);
    cur ^= 1;
  }

  // between iterations: hx = last, cx = 0
  k_prep2<<<(B_*H_+255)/256, 256, 0, stream>>>(lastf, hf, cf, hb0);
  cur = 0;

  // attention gate: h1 = tanh([inp, last] @ aW1^T + ab1); att = sigmoid(h1 @ aW2^T + ab2)
  k_attin<<<4096, 256, 0, stream>>>(inpb, hb0, attin);
  k_gemm<<<dim3(BT/64, ANP/128), 256, 0, stream>>>(attin, AKP, aW1b, AKP, ab1p, h1, ANP, AKP, 1);
  k_attreduce<<<BT, 32, 0, stream>>>(h1, aW2, ab2, attv);

  // iteration 2: gated LSTM scan
  for (int t = 0; t < T_; ++t){
    k_step<<<32, 128, 0, stream>>>(hb[cur], hb[cur^1], hf, cf, xg, Whhb,
                                   lastf, attv, len, t);
    cur ^= 1;
  }

  // logits = last @ Wout^T + bout   [64 x 5]
  k_out<<<1, 320, 0, stream>>>(lastf, Wout, bout, (float*)d_out);
}
